// SogCLR_DRO_Loss_46411416600720
// MI455X (gfx1250) — compile-verified, matrix operand order checked by reading, not run on hardware
//
#include <hip/hip_runtime.h>
#include <math.h>

#define BSZ 4096
#define DIM 768

#define GAMMA_   0.8f
#define TAU_MIN_ 0.005f
#define TAU_MAX_ 1.0f
#define RHO_     0.1f
#define ETA_INIT_ 0.01f
#define ETA_MIN_  0.0001f
#define BETA_U_  0.9f
#define EPS_     1e-10f
#define GCLIP_   5.0f

typedef __attribute__((ext_vector_type(16))) __bf16 v16b;
typedef __attribute__((ext_vector_type(8)))  __bf16 v8b;
typedef __attribute__((ext_vector_type(8)))  float  v8f;

// ---------------------------------------------------------------------------
// Prepack: split fp32 into bf16 hi + bf16 lo (x ~= hi + lo) once, so the GEMM
// inner loop is pure b128 loads + WMMA with no conversion VALU.
// ---------------------------------------------------------------------------
__global__ void prepack(const float* __restrict__ x, __bf16* __restrict__ hi,
                        __bf16* __restrict__ lo, int n) {
  int i = blockIdx.x * blockDim.x + threadIdx.x;
  if (i >= n) return;
  float v = x[i];
  __bf16 h = (__bf16)v;
  hi[i] = h;
  lo[i] = (__bf16)(v - (float)h);
}

// Load one 16x32 (MxK) bf16 fragment from a row-major bf16 matrix.
// Lane half h in {0,1}: elements 0..7  = K kBase+8h .. kBase+8h+7
//                       elements 8..15 = K kBase+16+8h .. kBase+16+8h+7
// -> two 16-byte (b128) loads per lane.
__device__ inline v16b load_fragb(const __bf16* __restrict__ base, int rc,
                                  int kBase, int half) {
  const __bf16* p = base + (size_t)rc * DIM + kBase + half * 8;
  v8b a = *(const v8b*)(p);
  v8b b = *(const v8b*)(p + 16);
  return __builtin_shufflevector(a, b, 0, 1, 2, 3, 4, 5, 6, 7,
                                 8, 9, 10, 11, 12, 13, 14, 15);
}

// ---------------------------------------------------------------------------
// sim = img @ txt^T via bf16x3 split WMMA (hi*hi + hi*lo + lo*hi ~ fp32).
// One wave owns a 32x64 output tile (2x4 fragments, 24 WMMAs / K=32 step).
// Block = 8 waves (2 in M x 4 in N) covering a 64(M) x 256(N) tile.
// ---------------------------------------------------------------------------
__global__ __launch_bounds__(256) void gemm_bf16x3(
    const __bf16* __restrict__ imgH, const __bf16* __restrict__ imgL,
    const __bf16* __restrict__ txtH, const __bf16* __restrict__ txtL,
    float* __restrict__ sim) {
  const int lane = threadIdx.x & 31;
  const int wave = threadIdx.x >> 5;
  const int half = lane >> 4;
  const int l16  = lane & 15;
  const int wm = wave >> 2;        // 0..1
  const int wn = wave & 3;         // 0..3
  const int tileM = blockIdx.y * 64 + wm * 32;
  const int tileN = blockIdx.x * 256 + wn * 64;

  v8f c[2][4] = {};
  for (int kb = 0; kb < DIM; kb += 32) {
    v16b ah[2], al[2], bh[4], bl[4];
#pragma unroll
    for (int mi = 0; mi < 2; ++mi) {
      const int r = tileM + mi * 16 + l16;
      ah[mi] = load_fragb(imgH, r, kb, half);
      al[mi] = load_fragb(imgL, r, kb, half);
    }
#pragma unroll
    for (int ni = 0; ni < 4; ++ni) {
      const int cq = tileN + ni * 16 + l16;
      bh[ni] = load_fragb(txtH, cq, kb, half);
      bl[ni] = load_fragb(txtL, cq, kb, half);
    }
#pragma unroll
    for (int mi = 0; mi < 2; ++mi)
#pragma unroll
      for (int ni = 0; ni < 4; ++ni) {
        c[mi][ni] = __builtin_amdgcn_wmma_f32_16x16x32_bf16(
            false, ah[mi], false, bh[ni], (short)0, c[mi][ni], false, false);
        c[mi][ni] = __builtin_amdgcn_wmma_f32_16x16x32_bf16(
            false, ah[mi], false, bl[ni], (short)0, c[mi][ni], false, false);
        c[mi][ni] = __builtin_amdgcn_wmma_f32_16x16x32_bf16(
            false, al[mi], false, bh[ni], (short)0, c[mi][ni], false, false);
      }
  }
  // C/D layout: VGPR e -> row (e + 8*half), col = lane&15 (within 16x16 tile)
#pragma unroll
  for (int mi = 0; mi < 2; ++mi)
#pragma unroll
    for (int ni = 0; ni < 4; ++ni) {
      const int col = tileN + ni * 16 + l16;
#pragma unroll
      for (int e = 0; e < 8; ++e) {
        const int row = tileM + mi * 16 + e + 8 * half;
        sim[(size_t)row * BSZ + col] = c[mi][ni][e];
      }
    }
}

__global__ void diag_kernel(const float* __restrict__ sim, float* __restrict__ diag) {
  int i = blockIdx.x * blockDim.x + threadIdx.x;
  if (i < BSZ) diag[i] = sim[(size_t)i * BSZ + i];
}

__device__ inline float eta_cur(int epoch, int max_epoch) {
  return ETA_MIN_ + (ETA_INIT_ - ETA_MIN_) *
         cosf(3.14159265358979f * ((float)epoch / (float)max_epoch) * 0.5f);
}

// One block per row i (image side). Two passes over the 16KB row (L2-hot):
// pass1 row max, pass2 exp sums; thread 0 finishes per-sample update.
// rs layout: [0]=s_new [B]=b_new [2B]=u_new [3B]=tau_new [4B]=loss [5B]=tau_old
__global__ __launch_bounds__(256) void row_pass(
    const float* __restrict__ sim, const float* __restrict__ diag,
    const float* __restrict__ s_I, const float* __restrict__ b_I,
    const float* __restrict__ u_I, const float* __restrict__ tau_I,
    const int* __restrict__ image_ids, const int* __restrict__ ep,
    const int* __restrict__ mep, float* __restrict__ rs) {
  const int i = blockIdx.x;
  const int t = threadIdx.x;
  const int id = image_ids[i];
  const float taui = tau_I[id];
  const float inv_tau = 1.0f / taui;
  const float d = diag[i];
  const float* row = sim + (size_t)i * BSZ;

  __shared__ float r0[256], r1[256], r2[256];

  float m = -INFINITY;
  for (int j = t; j < BSZ; j += 256) m = fmaxf(m, (row[j] - d) * inv_tau);
  r0[t] = m; __syncthreads();
  for (int s = 128; s > 0; s >>= 1) { if (t < s) r0[t] = fmaxf(r0[t], r0[t + s]); __syncthreads(); }
  const float old_b = b_I[id];
  const float new_b = fmaxf(r0[0], old_b);
  __syncthreads();

  float se = 0.f, swd = 0.f, swi = 0.f;
  for (int j = t; j < BSZ; j += 256) {
    float diff = row[j] - d;
    float idt  = diff * inv_tau;
    float e    = expf(idt - new_b);
    se += e; swd += e * diff; swi += e * idt;
  }
  r0[t] = se; r1[t] = swd; r2[t] = swi; __syncthreads();
  for (int s = 128; s > 0; s >>= 1) {
    if (t < s) { r0[t] += r0[t + s]; r1[t] += r1[t + s]; r2[t] += r2[t + s]; }
    __syncthreads();
  }
  if (t == 0) {
    const float g     = r0[0] / (float)(BSZ - 1);
    const float s_new = (1.f - GAMMA_) * s_I[id] * expf(old_b - new_b) + GAMMA_ * g;
    const float wsc   = 1.f / ((s_new + EPS_) * (float)(BSZ - 1));
    const float loss  = r1[0] * wsc;
    float grad = logf(s_new) + new_b + RHO_ - r2[0] * wsc;
    grad = fminf(fmaxf(grad, -GCLIP_), GCLIP_);
    const float u_new = (1.f - BETA_U_) * u_I[id] + BETA_U_ * grad;
    const float eta   = eta_cur(ep[0], mep[0]);
    const float tau_n = fminf(fmaxf(taui - eta * u_new, TAU_MIN_), TAU_MAX_);
    rs[i]            = s_new;
    rs[BSZ + i]      = new_b;
    rs[2 * BSZ + i]  = u_new;
    rs[3 * BSZ + i]  = tau_n;
    rs[4 * BSZ + i]  = loss;
    rs[5 * BSZ + i]  = taui;
  }
}

// Text side: one thread per column, 256 consecutive columns per block so row
// streaming is fully coalesced. Two passes (max, then sums).
__global__ __launch_bounds__(256) void col_pass(
    const float* __restrict__ sim, const float* __restrict__ diag,
    const float* __restrict__ s_T, const float* __restrict__ b_T,
    const float* __restrict__ u_T, const float* __restrict__ tau_T,
    const int* __restrict__ text_ids, const int* __restrict__ ep,
    const int* __restrict__ mep, float* __restrict__ cs) {
  const int c = blockIdx.x * 256 + threadIdx.x;
  if (c >= BSZ) return;
  const int id = text_ids[c];
  const float tauc = tau_T[id];
  const float inv_tau = 1.0f / tauc;
  const float d = diag[c];

  float m = -INFINITY;
  for (int r = 0; r < BSZ; ++r) m = fmaxf(m, (sim[(size_t)r * BSZ + c] - d) * inv_tau);
  const float old_b = b_T[id];
  const float new_b = fmaxf(m, old_b);

  float se = 0.f, swd = 0.f, swi = 0.f;
  for (int r = 0; r < BSZ; ++r) {
    float diff = sim[(size_t)r * BSZ + c] - d;
    float tdt  = diff * inv_tau;
    float e    = expf(tdt - new_b);
    se += e; swd += e * diff; swi += e * tdt;
  }
  const float g     = se / (float)(BSZ - 1);
  const float s_new = (1.f - GAMMA_) * s_T[id] * expf(old_b - new_b) + GAMMA_ * g;
  const float wsc   = 1.f / ((s_new + EPS_) * (float)(BSZ - 1));
  const float loss  = swd * wsc;
  float grad = logf(s_new) + new_b + RHO_ - swi * wsc;
  grad = fminf(fmaxf(grad, -GCLIP_), GCLIP_);
  const float u_new = (1.f - BETA_U_) * u_T[id] + BETA_U_ * grad;
  const float eta   = eta_cur(ep[0], mep[0]);
  const float tau_n = fminf(fmaxf(tauc - eta * u_new, TAU_MIN_), TAU_MAX_);
  cs[c]            = s_new;
  cs[BSZ + c]      = new_b;
  cs[2 * BSZ + c]  = u_new;
  cs[3 * BSZ + c]  = tau_n;
  cs[4 * BSZ + c]  = loss;
  cs[5 * BSZ + c]  = tauc;
}

// Scalars: total_loss, tau_img.mean, tau_txt.mean
__global__ __launch_bounds__(256) void finalize(const float* __restrict__ rs,
                                                const float* __restrict__ cs,
                                                float* __restrict__ out, int N) {
  const int t = threadIdx.x;
  __shared__ float a0[256], a1[256], a2[256], a3[256];
  float il = 0.f, tl = 0.f, ti = 0.f, tt = 0.f;
  for (int i = t; i < BSZ; i += 256) {
    il += rs[4 * BSZ + i]; tl += cs[4 * BSZ + i];
    ti += rs[5 * BSZ + i]; tt += cs[5 * BSZ + i];
  }
  a0[t] = il; a1[t] = tl; a2[t] = ti; a3[t] = tt; __syncthreads();
  for (int s = 128; s > 0; s >>= 1) {
    if (t < s) { a0[t] += a0[t + s]; a1[t] += a1[t + s]; a2[t] += a2[t + s]; a3[t] += a3[t + s]; }
    __syncthreads();
  }
  if (t == 0) {
    out[0]                     = a0[0] / (float)BSZ + a1[0] / (float)BSZ;
    out[1 + 8 * (size_t)N]     = a2[0] / (float)BSZ;
    out[1 + 8 * (size_t)N + 1] = a3[0] / (float)BSZ;
  }
}

// Copy the 8 big state buffers into the output (before scatter).
__global__ void copy_state(const float* __restrict__ s_I, const float* __restrict__ s_T,
                           const float* __restrict__ b_I, const float* __restrict__ b_T,
                           const float* __restrict__ u_I, const float* __restrict__ u_T,
                           const float* __restrict__ tau_I, const float* __restrict__ tau_T,
                           float* __restrict__ out, int N) {
  int i = blockIdx.x * blockDim.x + threadIdx.x;
  if (i >= N) return;
  float* o = out + 1;
  o[i]                 = s_I[i];
  o[(size_t)N + i]     = s_T[i];
  o[2 * (size_t)N + i] = b_I[i];
  o[3 * (size_t)N + i] = b_T[i];
  o[4 * (size_t)N + i] = u_I[i];
  o[5 * (size_t)N + i] = u_T[i];
  o[6 * (size_t)N + i] = tau_I[i];
  o[7 * (size_t)N + i] = tau_T[i];
}

__global__ void scatter_updates(const float* __restrict__ rs, const float* __restrict__ cs,
                                const int* __restrict__ image_ids,
                                const int* __restrict__ text_ids,
                                float* __restrict__ out, int N) {
  int i = blockIdx.x * blockDim.x + threadIdx.x;
  if (i >= BSZ) return;
  float* o = out + 1;
  const int ii = image_ids[i];
  const int it = text_ids[i];
  o[ii]                 = rs[i];            // s_I
  o[(size_t)N + it]     = cs[i];            // s_T
  o[2 * (size_t)N + ii] = rs[BSZ + i];      // b_I
  o[3 * (size_t)N + it] = cs[BSZ + i];      // b_T
  o[4 * (size_t)N + ii] = rs[2 * BSZ + i];  // u_I
  o[5 * (size_t)N + it] = cs[2 * BSZ + i];  // u_T
  o[6 * (size_t)N + ii] = rs[3 * BSZ + i];  // tau_I
  o[7 * (size_t)N + it] = cs[3 * BSZ + i];  // tau_T
}

extern "C" void kernel_launch(void* const* d_in, const int* in_sizes, int n_in,
                              void* d_out, int out_size, void* d_ws, size_t ws_size,
                              hipStream_t stream) {
  const float* img     = (const float*)d_in[0];
  const float* txt     = (const float*)d_in[1];
  const float* s_I     = (const float*)d_in[2];
  const float* s_T     = (const float*)d_in[3];
  const float* b_I     = (const float*)d_in[4];
  const float* b_T     = (const float*)d_in[5];
  const float* u_I     = (const float*)d_in[6];
  const float* u_T     = (const float*)d_in[7];
  const float* tau_I   = (const float*)d_in[8];
  const float* tau_T   = (const float*)d_in[9];
  const int* image_ids = (const int*)d_in[10];
  const int* text_ids  = (const int*)d_in[11];
  const int* epoch     = (const int*)d_in[12];
  const int* max_epoch = (const int*)d_in[13];
  const int N = in_sizes[2];  // N_STATE
  float* out = (float*)d_out;

  // Workspace layout (floats first, then bf16 prepack arrays; all 16B aligned)
  float* sim  = (float*)d_ws;                       // BSZ*BSZ
  float* diag = sim + (size_t)BSZ * BSZ;            // BSZ
  float* rs   = diag + BSZ;                         // 6*BSZ
  float* cs   = rs + 6 * BSZ;                       // 6*BSZ
  __bf16* imgH = (__bf16*)(cs + 6 * BSZ);           // BSZ*DIM each
  __bf16* imgL = imgH + (size_t)BSZ * DIM;
  __bf16* txtH = imgL + (size_t)BSZ * DIM;
  __bf16* txtL = txtH + (size_t)BSZ * DIM;

  const int nElem = BSZ * DIM;
  prepack<<<(nElem + 255) / 256, 256, 0, stream>>>(img, imgH, imgL, nElem);
  prepack<<<(nElem + 255) / 256, 256, 0, stream>>>(txt, txtH, txtL, nElem);

  dim3 gGrid(BSZ / 256, BSZ / 64);
  gemm_bf16x3<<<gGrid, 256, 0, stream>>>(imgH, imgL, txtH, txtL, sim);
  diag_kernel<<<BSZ / 256, 256, 0, stream>>>(sim, diag);
  row_pass<<<BSZ, 256, 0, stream>>>(sim, diag, s_I, b_I, u_I, tau_I,
                                    image_ids, epoch, max_epoch, rs);
  col_pass<<<BSZ / 256, 256, 0, stream>>>(sim, diag, s_T, b_T, u_T, tau_T,
                                          text_ids, epoch, max_epoch, cs);
  finalize<<<1, 256, 0, stream>>>(rs, cs, out, N);
  copy_state<<<(N + 255) / 256, 256, 0, stream>>>(s_I, s_T, b_I, b_T, u_I, u_T,
                                                  tau_I, tau_T, out, N);
  scatter_updates<<<(BSZ + 255) / 256, 256, 0, stream>>>(rs, cs, image_ids, text_ids, out, N);
}
